// LabelPropagationMLP_83897891160439
// MI455X (gfx1250) — matrix operand order repaired and checked by reading, not compiled, not run on hardware
//
#include <hip/hip_runtime.h>
#include <hip/hip_bf16.h>

typedef __attribute__((ext_vector_type(16))) __bf16 v16bf;
typedef __attribute__((ext_vector_type(8)))  float  v8f;

#define H_DIM 256
#define C_DIM 64
#define EF_DIM 20

__device__ __forceinline__ float gelu_f(float x) {
  return 0.5f * x * (1.0f + tanhf(0.7978845608028654f * (x + 0.044715f * x * x * x)));
}

__device__ __forceinline__ void atomicMaxFloat(float* addr, float value) {
  if (value >= 0.0f)
    atomicMax((int*)addr, __float_as_int(value));
  else
    atomicMin((unsigned int*)addr, __float_as_uint(value));
}

// LDS byte offset of a flat shared pointer (ISA 10.2: low 32 bits of the address)
__device__ __forceinline__ unsigned lds_off(const void* p) {
  return (unsigned)(size_t)p;
}

// CDNA5 async global->LDS 16B DMA (tracked by ASYNCcnt)
__device__ __forceinline__ void async_g2l_b128(unsigned ldsaddr, const void* gaddr) {
  asm volatile("global_load_async_to_lds_b128 %0, %1, off"
               :: "v"(ldsaddr), "v"(gaddr) : "memory");
}
__device__ __forceinline__ void wait_asynccnt0() {
  asm volatile("s_wait_asynccnt 0x0" ::: "memory");
}

// ---------------------------------------------------------------- fp32 -> bf16
__global__ void k_cvt_bf16(const float* __restrict__ in, __bf16* __restrict__ out, long n) {
  long i = (long)blockIdx.x * blockDim.x + threadIdx.x;
  if (i < n) out[i] = (__bf16)in[i];
}

// fp32 [K,Nc] -> bf16 transposed [Nc,K] (weights only, one-time, tiny)
__global__ void k_cvt_bf16_t(const float* __restrict__ in, __bf16* __restrict__ out,
                             int K, int Nc) {
  int i = blockIdx.x * blockDim.x + threadIdx.x;
  if (i >= K * Nc) return;
  int n = i / K, k = i - n * K;
  out[(size_t)n * K + k] = (__bf16)in[(size_t)k * Nc + n];
}

// ---------------------------------------------------------------- WMMA GEMM
// C[M,Nc] = epilogue(A[M,K] @ Bt^T + bias), A row-major [M,K], Bt row-major [Nc,K].
// EPI==0: gelu -> bf16 out. EPI==1: identity -> f32 out.
// Block: 128 threads (4 waves). Block tile 64x64. Wave w owns rows [16w,16w+16).
union FragU { v16bf v; uint4 u[2]; };

template <int EPI>
__global__ void __launch_bounds__(128)
k_gemm_bf16(const __bf16* __restrict__ A, const __bf16* __restrict__ Bt,
            const float* __restrict__ bias, void* __restrict__ Out,
            int M, int K, int Nc) {
  constexpr int LDT = 40;  // padded row stride in bf16 (80B: 16B-aligned, bank-staggered)
  __shared__ __align__(16) __bf16 sA[64][LDT];   // [m][k]
  __shared__ __align__(16) __bf16 sB[64][LDT];   // [n][k]

  const int tid  = threadIdx.x;
  const int lane = tid & 31;
  const int wv   = tid >> 5;
  const int tileN = blockIdx.x * 64;   // x = N tiles: adjacent blocks share the A strip in L2
  const int tileM = blockIdx.y * 64;

  v8f acc[4] = {v8f{}, v8f{}, v8f{}, v8f{}};

  // staging assignment: thread -> (row 0..63, 16-elem k segment)
  const int srow = tid >> 1;
  const int sseg = (tid & 1) * 16;
  int gm = tileM + srow;
  if (gm >= M) gm = M - 1;             // clamp; OOB rows masked at the store
  const __bf16* gA = A  + (size_t)gm * K + sseg;
  const __bf16* gB = Bt + (size_t)(tileN + srow) * K + sseg;
  const unsigned la = lds_off(&sA[srow][sseg]);
  const unsigned lb = lds_off(&sB[srow][sseg]);

  for (int kb = 0; kb < K; kb += 32) {
    __syncthreads();
    // async DMA: 8 bf16 of this row's K-panel into LDS (1x16B each for A and B)
    async_g2l_b128(la, gA + kb);
    async_g2l_b128(lb, gB + kb);
    wait_asynccnt0();
    __syncthreads();

    // A fragment (16x32 bf16): lane m = lane%16; K pattern per ISA layout
    FragU af;
    {
      int m  = wv * 16 + (lane & 15);
      int kh = (lane >> 4) * 8;
      af.u[0] = *(const uint4*)&sA[m][kh];       // K = kh .. kh+7
      af.u[1] = *(const uint4*)&sA[m][kh + 16];  // K = 16+kh .. 16+kh+7
    }
    const int nl = lane & 15;
  const int k0 = (lane >> 4) * 8;
    #pragma unroll
    for (int ns = 0; ns < 4; ++ns) {
      FragU bf;
      bf.u[0] = *(const uint4*)&sB[ns * 16 + nl][k0];
      bf.u[1] = *(const uint4*)&sB[ns * 16 + nl][k0 + 16];
      acc[ns] = __builtin_amdgcn_wmma_f32_16x16x32_bf16(
          false, af.v, false, bf.v, (short)0, acc[ns], false, false);
    }
  }

  // epilogue: C/D layout: reg r -> M = 16*wv + 8*(lane/16) + r, N = lane%16
  const int half = lane >> 4;
  const int nlc  = lane & 15;
  #pragma unroll
  for (int ns = 0; ns < 4; ++ns) {
    int col = tileN + ns * 16 + nlc;
    float bv = bias[col];
    #pragma unroll
    for (int r = 0; r < 8; ++r) {
      int row = tileM + wv * 16 + half * 8 + r;
      if (row < M) {
        float v = acc[ns][r] + bv;
        if (EPI == 0)
          ((__bf16*)Out)[(size_t)row * Nc + col] = (__bf16)gelu_f(v);
        else
          ((float*)Out)[(size_t)row * Nc + col] = v;
      }
    }
  }
}

// ---------------------------------------------------------------- LayerNorm (row=256)
__global__ void __launch_bounds__(256)
k_layernorm(const float* __restrict__ hin, const float* __restrict__ g,
            const float* __restrict__ b, __bf16* __restrict__ hout) {
  int row = blockIdx.x;
  int tid = threadIdx.x;
  float x = hin[(size_t)row * H_DIM + tid];
  float s1 = x, s2 = x * x;
  #pragma unroll
  for (int off = 16; off; off >>= 1) {
    s1 += __shfl_down(s1, off);
    s2 += __shfl_down(s2, off);
  }
  __shared__ float w1[8], w2[8];
  __shared__ float mu_s, rs_s;
  int lane = tid & 31, wv = tid >> 5;
  if (lane == 0) { w1[wv] = s1; w2[wv] = s2; }
  __syncthreads();
  if (tid == 0) {
    float a = 0.f, c = 0.f;
    #pragma unroll
    for (int i = 0; i < 8; ++i) { a += w1[i]; c += w2[i]; }
    float mu  = a * (1.0f / 256.0f);
    float var = c * (1.0f / 256.0f) - mu * mu;
    mu_s = mu; rs_s = rsqrtf(var + 1e-5f);
  }
  __syncthreads();
  hout[(size_t)row * H_DIM + tid] = (__bf16)((x - mu_s) * rs_s * g[tid] + b[tid]);
}

// ---------------------------------------------------------------- k2 per edge type (ET tiny)
__global__ void k_k2types(const float* __restrict__ emb_table, const float* __restrict__ We,
                          const float* __restrict__ be, float* __restrict__ k2t) {
  int t = blockIdx.x, h = threadIdx.x;
  float acc = be[h];
  #pragma unroll
  for (int f = 0; f < EF_DIM; ++f)
    acc += gelu_f(emb_table[t * EF_DIM + f]) * We[f * H_DIM + h];
  k2t[t * H_DIM + h] = acc;
}

// ---------------------------------------------------------------- init out / segment state
__global__ void k_init(float* __restrict__ out, float* __restrict__ nmax,
                       float* __restrict__ nsum, long nOut, int Nn) {
  long i = (long)blockIdx.x * blockDim.x + threadIdx.x;
  if (i < nOut) out[i] = 0.0f;
  if (i < Nn) { nmax[i] = -__builtin_huge_valf(); nsum[i] = 0.0f; }
}

// ---------------------------------------------------------------- per-edge alpha + segment max
__global__ void __launch_bounds__(256)
k_alpha(const float* __restrict__ q, const float* __restrict__ k1,
        const float* __restrict__ k2t, const int* __restrict__ src,
        const int* __restrict__ dst, const int* __restrict__ typ,
        float* __restrict__ alpha, float* __restrict__ nmax, int E) {
  int e = blockIdx.x * 8 + (threadIdx.x >> 5);
  if (e >= E) return;
  int lane = threadIdx.x & 31;
  int s = src[e], d = dst[e], t = typ[e];
  const float4* qp = (const float4*)(q + (size_t)d * H_DIM);
  const float4* kp = (const float4*)(k1 + (size_t)s * H_DIM);
  const float4* tp = (const float4*)(k2t + (size_t)t * H_DIM);
  float acc = 0.0f;
  #pragma unroll
  for (int i = 0; i < 2; ++i) {
    int idx = lane + i * 32;  // 64 float4 per row
    float4 qv = qp[idx], kv = kp[idx], tv = tp[idx];
    acc += qv.x * (kv.x + tv.x) + qv.y * (kv.y + tv.y) +
           qv.z * (kv.z + tv.z) + qv.w * (kv.w + tv.w);
  }
  #pragma unroll
  for (int off = 16; off; off >>= 1) acc += __shfl_down(acc, off);
  if (lane == 0) {
    float a = acc * 0.0625f;  // / sqrt(256)
    alpha[e] = a;
    atomicMaxFloat(&nmax[d], a);
  }
}

// ---------------------------------------------------------------- exp + segment sum
__global__ void k_expsum(float* __restrict__ alp, const float* __restrict__ nmx,
                         float* __restrict__ nsm, const int* __restrict__ dst, int E) {
  int e = blockIdx.x * blockDim.x + threadIdx.x;
  if (e >= E) return;
  int d = dst[e];
  float v = __expf(alp[e] - nmx[d]);
  alp[e] = v;
  atomicAdd(&nsm[d], v);
}

// ---------------------------------------------------------------- weighted scatter of scores
__global__ void __launch_bounds__(256)
k_aggregate(const float* __restrict__ alp, const float* __restrict__ nsm,
            const float* __restrict__ scores, const int* __restrict__ src,
            const int* __restrict__ dst, float* __restrict__ out, int E) {
  int e = blockIdx.x * 8 + (threadIdx.x >> 5);
  if (e >= E) return;
  int lane = threadIdx.x & 31;
  int s = src[e], d = dst[e];
  float coef = alp[e] / nsm[d];
  const float2* sp = (const float2*)(scores + (size_t)s * C_DIM);
  float2 v = sp[lane];
  atomicAdd(&out[(size_t)d * C_DIM + lane * 2 + 0], v.x * coef);
  atomicAdd(&out[(size_t)d * C_DIM + lane * 2 + 1], v.y * coef);
}

// ================================================================ launch
extern "C" void kernel_launch(void* const* d_in, const int* in_sizes, int n_in,
                              void* d_out, int out_size, void* d_ws, size_t ws_size,
                              hipStream_t stream) {
  const float* embedding = (const float*)d_in[0];
  const float* scores    = (const float*)d_in[1];
  const int*   src       = (const int*)d_in[2];
  const int*   dst       = (const int*)d_in[3];
  const int*   typ       = (const int*)d_in[4];
  const float* W1        = (const float*)d_in[5];
  const float* b1        = (const float*)d_in[6];
  const float* W2        = (const float*)d_in[7];
  const float* b2        = (const float*)d_in[8];
  const float* ln_g      = (const float*)d_in[9];
  const float* ln_b      = (const float*)d_in[10];
  const float* Wl        = (const float*)d_in[11];
  const float* bl        = (const float*)d_in[12];
  const float* Wr        = (const float*)d_in[13];
  const float* br        = (const float*)d_in[14];
  const float* We        = (const float*)d_in[15];
  const float* be        = (const float*)d_in[16];
  const float* emb_table = (const float*)d_in[17];

  const int H  = H_DIM, C = C_DIM, EF = EF_DIM;
  const int Nn = in_sizes[0] / H;
  const int E  = in_sizes[2];
  const int ET = in_sizes[17] / EF;
  const int H4 = 4 * H;

  char* base = (char*)d_ws;
  size_t off = 0;
  auto alloc = [&](size_t bytes) -> char* {
    char* p = base + off;
    off = (off + bytes + 255) & ~(size_t)255;
    return p;
  };
  __bf16* Xb  = (__bf16*)alloc((size_t)Nn * H * 2);
  __bf16* W1b = (__bf16*)alloc((size_t)H * H4 * 2);   // [H4, H]   (transposed)
  __bf16* W2b = (__bf16*)alloc((size_t)H4 * H * 2);   // [H, H4]   (transposed)
  __bf16* Wlb = (__bf16*)alloc((size_t)H * H * 2);    // [H, H]    (transposed)
  __bf16* Wrb = (__bf16*)alloc((size_t)H * H * 2);    // [H, H]    (transposed)
  __bf16* A1b = (__bf16*)alloc((size_t)Nn * H4 * 2);
  __bf16* Hb  = (__bf16*)alloc((size_t)Nn * H * 2);
  float*  qb  = (float*)alloc((size_t)Nn * H * 4);    // pre-LN h, then reused for q
  float*  k1b = (float*)alloc((size_t)Nn * H * 4);
  float*  k2t = (float*)alloc((size_t)ET * H * 4);
  float*  alp = (float*)alloc((size_t)E * 4);
  float*  nmx = (float*)alloc((size_t)Nn * 4);
  float*  nsm = (float*)alloc((size_t)Nn * 4);
  (void)ws_size; (void)n_in; (void)out_size;

  // activation convert (layout kept), weights convert+transpose (one-time)
  long nx = (long)Nn * H;
  k_cvt_bf16<<<(unsigned)((nx + 255) / 256), 256, 0, stream>>>(embedding, Xb, nx);
  k_cvt_bf16_t<<<(H * H4 + 255) / 256, 256, 0, stream>>>(W1, W1b, H, H4);
  k_cvt_bf16_t<<<(H4 * H + 255) / 256, 256, 0, stream>>>(W2, W2b, H4, H);
  k_cvt_bf16_t<<<(H * H + 255) / 256, 256, 0, stream>>>(Wl, Wlb, H, H);
  k_cvt_bf16_t<<<(H * H + 255) / 256, 256, 0, stream>>>(Wr, Wrb, H, H);

  const int Mt = (Nn + 63) / 64;
  // GEMM1: A1 = gelu(X @ W1 + b1)            [Nn,1024] bf16
  k_gemm_bf16<0><<<dim3(H4 / 64, Mt), 128, 0, stream>>>(Xb, W1b, b1, A1b, Nn, H, H4);
  // GEMM2: h_pre = A1 @ W2 + b2              [Nn,256] f32 (into qb)
  k_gemm_bf16<1><<<dim3(H / 64, Mt), 128, 0, stream>>>(A1b, W2b, b2, qb, Nn, H4, H);
  // LayerNorm -> Hb (bf16)
  k_layernorm<<<Nn, 256, 0, stream>>>(qb, ln_g, ln_b, Hb);
  // GEMM3: q  = h @ Wl + bl                  [Nn,256] f32
  k_gemm_bf16<1><<<dim3(H / 64, Mt), 128, 0, stream>>>(Hb, Wlb, bl, qb, Nn, H, H);
  // GEMM4: k1 = h @ Wr + br                  [Nn,256] f32
  k_gemm_bf16<1><<<dim3(H / 64, Mt), 128, 0, stream>>>(Hb, Wrb, br, k1b, Nn, H, H);
  // k2 per type (ET=3)
  k_k2types<<<ET, H, 0, stream>>>(emb_table, We, be, k2t);
  // init output + segment buffers
  long nOut = (long)Nn * C;
  k_init<<<(unsigned)((nOut + 255) / 256), 256, 0, stream>>>((float*)d_out, nmx, nsm, nOut, Nn);
  // edge phase
  int eb = (E + 7) / 8;
  k_alpha<<<eb, 256, 0, stream>>>(qb, k1b, k2t, src, dst, typ, alp, nmx, E);
  k_expsum<<<(E + 255) / 256, 256, 0, stream>>>(alp, nmx, nsm, dst, E);
  k_aggregate<<<eb, 256, 0, stream>>>(alp, nsm, scores, src, dst, (float*)d_out, E);
}
